// AttentionBlock_38654705664345
// MI455X (gfx1250) — matrix operand order repaired and checked by reading, not compiled, not run on hardware
//
#include <hip/hip_runtime.h>
#include <stdint.h>

// ---------------------------------------------------------------------------
// Types for CDNA5 WMMA (wave32, 16x16x32 bf16 -> f32)
// ---------------------------------------------------------------------------
typedef __attribute__((ext_vector_type(16))) __bf16 v16bf;
typedef __attribute__((ext_vector_type(8)))  float  v8f;

union BF16x16 { v16bf v; uint32_t u[8]; };

#define WMMA_BF16(a, b, c) \
  __builtin_amdgcn_wmma_f32_16x16x32_bf16(false, (a), false, (b), (short)0, (c), false, false)

__device__ __forceinline__ uint16_t f2bf(float f) {
  uint32_t x = __float_as_uint(f);
  return (uint16_t)((x + 0x7FFFu + ((x >> 16) & 1u)) >> 16); // RNE truncate
}
__device__ __forceinline__ v8f v8zero() {
  v8f z = {0.f,0.f,0.f,0.f,0.f,0.f,0.f,0.f};
  return z;
}

// CDNA5 async global->LDS copy (ASYNCcnt path; no VGPR round trip).
// LDS destination address = low 32 bits of the generic shared-aperture address.
__device__ __forceinline__ void async_g2l_b128(void* lds, const void* g) {
  uint32_t l = (uint32_t)(uintptr_t)lds;
  uint64_t a = (uint64_t)(uintptr_t)g;
  asm volatile("global_load_async_to_lds_b128 %0, %1, off" :: "v"(l), "v"(a) : "memory");
}
__device__ __forceinline__ void wait_async0() {
  asm volatile("s_wait_asynccnt 0x0" ::: "memory");
}

// ---------------------------------------------------------------------------
// Problem sizes
// ---------------------------------------------------------------------------
#define NB 8
#define NH 8
#define SCALE_QK 0.35355339059327373f  // 64^(-1/4)

// workspace byte offsets (total ~44.2 MB; resident in 192 MB L2)
#define OFF_MOD 0u                 // fp32 [8][1536]
#define OFF_WQ  65536u             // bf16 [1536][512]
#define OFF_WP  1638400u           // bf16 [512][512]
#define OFF_XMT 2162688u           // bf16 [8][1024][512]   (t-major, channel contiguous)
#define OFF_QT  10551296u          // bf16 [64][1024][64]   (per head, t-major, q*sc)
#define OFF_KT  18939904u          // bf16 [64][1024][64]   (per head, t-major, k*sc)
#define OFF_V   27328512u          // bf16 [64][64][1024]   (per head, ch-major)
#define OFF_AT  35717120u          // bf16 [8][1024][512]   (attention out, t-major)

// ---------------------------------------------------------------------------
// Kernel 0: fp32 -> bf16 weight conversion
// ---------------------------------------------------------------------------
__global__ void k_convert(const float* __restrict__ qkv_w, const float* __restrict__ proj_w,
                          uint16_t* __restrict__ Wq, uint16_t* __restrict__ Wp) {
  int i = blockIdx.x * 256 + threadIdx.x;
  if (i < 1536 * 512) {
    Wq[i] = f2bf(qkv_w[i]);
  } else {
    int j = i - 1536 * 512;
    if (j < 512 * 512) Wp[j] = f2bf(proj_w[j]);
  }
}

// ---------------------------------------------------------------------------
// Kernel 1: mod = silu(emb) @ ada_w^T + ada_b   ([8][1536], tiny GEMM)
// ---------------------------------------------------------------------------
__global__ void k_mod(const float* __restrict__ emb, const float* __restrict__ ada_w,
                      const float* __restrict__ ada_b, float* __restrict__ mod) {
  __shared__ float se[512];
  const int b = blockIdx.x, tid = threadIdx.x;
  for (int i = tid; i < 512; i += 256) {
    float e = emb[b * 512 + i];
    se[i] = e / (1.0f + __expf(-e));   // silu
  }
  __syncthreads();
  for (int o = tid; o < 1536; o += 256) {
    const float* wr = ada_w + (size_t)o * 512;
    float acc = ada_b[o];
    for (int k = 0; k < 512; ++k) acc += se[k] * wr[k];
    mod[b * 1536 + o] = acc;
  }
}

// ---------------------------------------------------------------------------
// Kernel 2: GroupNorm(32) + adaLN modulation -> transposed bf16 XMt[b][t][c]
// ---------------------------------------------------------------------------
__global__ void k_gn(const float* __restrict__ x, const float* __restrict__ mod,
                     uint16_t* __restrict__ XMt) {
  __shared__ float r1[256], r2[256];
  const int g = blockIdx.x, b = blockIdx.y, tid = threadIdx.x;
  const size_t base = ((size_t)b * 512 + g * 16) * 1024;
  float s = 0.f, s2 = 0.f;
  for (int i = tid; i < 16384; i += 256) {
    float xv = x[base + i];
    s += xv; s2 += xv * xv;
  }
  r1[tid] = s; r2[tid] = s2;
  __syncthreads();
  for (int st = 128; st > 0; st >>= 1) {
    if (tid < st) { r1[tid] += r1[tid + st]; r2[tid] += r2[tid + st]; }
    __syncthreads();
  }
  const float mu   = r1[0] * (1.0f / 16384.0f);
  const float var  = r2[0] * (1.0f / 16384.0f) - mu * mu;
  const float rinv = rsqrtf(var + 1e-5f);
  for (int i = tid; i < 16384; i += 256) {
    int c = g * 16 + (i >> 10), t = i & 1023;
    float sc = mod[b * 1536 + 512 + c];
    float sh = mod[b * 1536 + c];
    float xm = (x[base + i] - mu) * rinv * (1.0f + sc) + sh;
    XMt[((size_t)b * 1024 + t) * 512 + c] = f2bf(xm);
  }
}

// ---------------------------------------------------------------------------
// Kernel 3: QKV GEMM  Y[1536,1024] = Wq[1536,512] @ Xm[512,1024]  (per batch)
// B-tile (shared by all 8 waves) staged in LDS via async copies per K chunk.
// ---------------------------------------------------------------------------
__global__ void __launch_bounds__(256)
k_qkv(const uint16_t* __restrict__ Wq, const uint16_t* __restrict__ XMt,
      const float* __restrict__ qkv_b,
      uint16_t* __restrict__ Qt, uint16_t* __restrict__ Kt, uint16_t* __restrict__ Vm) {
  __shared__ uint16_t Xs[128][40];                  // [t-local][32 bf16 + pad], 10 KB
  uint32_t* Xs32 = (uint32_t*)&Xs[0][0];            // row stride 20 dwords (bank-injective)
  const int b = blockIdx.z;
  const int wave = threadIdx.x >> 5, lane = threadIdx.x & 31;
  const int m = lane & 15, hg = lane >> 4;
  const int obase = blockIdx.y * 128 + wave * 16;
  const int tbase = blockIdx.x * 128;
  const uint32_t* wrow = (const uint32_t*)(Wq + (size_t)(obase + m) * 512);
  const uint4* XG4 = (const uint4*)(XMt + ((size_t)b * 1024 + tbase) * 512); // row = 64 uint4

  v8f acc[8];
  #pragma unroll
  for (int ct = 0; ct < 8; ++ct) acc[ct] = v8zero();

  const int row0 = threadIdx.x >> 2, q0 = threadIdx.x & 3;  // staging slot

  for (int kk = 0; kk < 512; kk += 32) {
    __syncthreads();
    #pragma unroll
    for (int it = 0; it < 2; ++it) {                // stage 128x32 B tile (512 uint4)
      const int row = row0 + it * 64, q = q0;
      async_g2l_b128(&Xs32[row * 20 + q * 4], &XG4[row * 64 + (kk >> 3) + q]);
    }
    const int kpf = (kk + 32 < 512) ? (kk + 32) : kk;        // clamped, branch-free
    __builtin_prefetch((const void*)&XG4[row0 * 64 + (kpf >> 3) + q0], 0, 1);
    wait_async0();
    __syncthreads();

    BF16x16 a;
    #pragma unroll
    for (int j = 0; j < 8; ++j)
      a.u[j] = wrow[(kk >> 1) + hg * 8 + j];   // A layout (global)
    #pragma unroll
    for (int ct = 0; ct < 8; ++ct) {
      BF16x16 bb;
      #pragma unroll
      for (int j = 0; j < 8; ++j)
        bb.u[j] = Xs32[(ct * 16 + m) * 20 + hg * 8 + j];            // B layout (LDS)
      acc[ct] = WMMA_BF16(a.v, bb.v, acc[ct]);
    }
  }

  #pragma unroll
  for (int ct = 0; ct < 8; ++ct) {
    const int t = tbase + ct * 16 + m;
    #pragma unroll
    for (int v = 0; v < 8; ++v) {
      const int o = obase + v + 8 * hg;            // C layout row
      float val = acc[ct][v] + qkv_b[o];
      const int head = o / 192, r = o % 192, kind = r >> 6, c = r & 63;
      const size_t bh = (size_t)(b * 8 + head);
      if (kind == 0)      Qt[bh * 65536 + (size_t)t * 64 + c]   = f2bf(val * SCALE_QK);
      else if (kind == 1) Kt[bh * 65536 + (size_t)t * 64 + c]   = f2bf(val * SCALE_QK);
      else                Vm[bh * 65536 + (size_t)c * 1024 + t] = f2bf(val);
    }
  }
}

// ---------------------------------------------------------------------------
// Kernel 4: flash attention per head. 4 waves x 16 query rows per workgroup.
// K/V 128-key blocks async-staged in LDS (shared by all waves); online softmax;
// P routed through padded LDS tile to re-materialize the WMMA A-layout.
// ---------------------------------------------------------------------------
__global__ void __launch_bounds__(128)
k_attn(const uint16_t* __restrict__ Qt, const uint16_t* __restrict__ Kt,
       const uint16_t* __restrict__ Vm, uint16_t* __restrict__ At) {
  __shared__ uint16_t Ks[128][72];    // [tk-local][64 bf16 + pad]  18.0 KB
  __shared__ uint16_t Vs[64][136];    // [cout][128 bf16 + pad]     17.0 KB
  __shared__ uint16_t Pl[4][16][132]; // per-wave P tile, padded    16.5 KB
  uint32_t* Ks32 = (uint32_t*)&Ks[0][0];   // row stride 36 dwords
  uint32_t* Vs32 = (uint32_t*)&Vs[0][0];   // row stride 68 dwords

  const int bh   = blockIdx.y;                        // b*8 + head
  const int wave = threadIdx.x >> 5, lane = threadIdx.x & 31;
  const int m = lane & 15, hg = lane >> 4;
  const int qrow0 = blockIdx.x * 64 + wave * 16;

  // Q tile as A-operands for c=0..31 and c=32..63 (registers, reused all blocks)
  const uint32_t* qp = (const uint32_t*)(Qt + (size_t)bh * 65536 + (size_t)(qrow0 + m) * 64);
  BF16x16 aq0, aq1;
  #pragma unroll
  for (int j = 0; j < 8; ++j) {
    const int idx = hg * 8 + j;
    aq0.u[j] = qp[idx];
    aq1.u[j] = qp[16 + idx];
  }

  float mrun[8], lrun[8];
  v8f oacc[4];
  #pragma unroll
  for (int v = 0; v < 8; ++v) { mrun[v] = -1e30f; lrun[v] = 0.0f; }
  #pragma unroll
  for (int i = 0; i < 4; ++i) oacc[i] = v8zero();

  const uint4* KG4 = (const uint4*)(Kt + (size_t)bh * 65536);  // [1024][8 uint4]
  const uint4* VG4 = (const uint4*)(Vm + (size_t)bh * 65536);  // [64][128 uint4]

  for (int kb = 0; kb < 8; ++kb) {
    const int tk0 = kb * 128;

    __syncthreads();   // previous block's LDS reads complete
    // ---- async-stage K block [128][64] and V block [64][128] (1024 uint4 each) ----
    #pragma unroll
    for (int it = 0; it < 8; ++it) {
      const int idx = threadIdx.x + it * 128;
      { const int row = idx >> 3, q = idx & 7;       // K: 8 uint4 per row
        async_g2l_b128(&Ks32[row * 36 + q * 4], &KG4[(tk0 + row) * 8 + q]); }
      { const int row = idx >> 4, q = idx & 15;      // V: 16 uint4 per 128-key slice
        async_g2l_b128(&Vs32[row * 68 + q * 4], &VG4[row * 128 + (tk0 >> 3) + q]); }
    }
    {
      const int tnext = (kb + 1 < 8) ? (tk0 + 128) : tk0;      // clamped, branch-free
      const int row = (threadIdx.x & 63) << 1;
      __builtin_prefetch((const void*)&KG4[(tnext + row) * 8], 0, 1);
      __builtin_prefetch((const void*)&VG4[row * 128 + (tnext >> 3)], 0, 1);
    }
    wait_async0();
    __syncthreads();

    // ---- S block [16 x 128] from LDS K ----
    v8f s[8];
    #pragma unroll
    for (int ct = 0; ct < 8; ++ct) {
      const uint32_t* kp = Ks32 + (size_t)(ct * 16 + m) * 36;   // column N = key
      BF16x16 b0, b1;
      #pragma unroll
      for (int j = 0; j < 8; ++j) { b0.u[j] = kp[hg * 8 + j]; b1.u[j] = kp[16 + hg * 8 + j]; }
      v8f z = v8zero();
      z = WMMA_BF16(aq0.v, b0.v, z);
      z = WMMA_BF16(aq1.v, b1.v, z);
      s[ct] = z;
    }

    // ---- online softmax: block row max (across 16 lanes of half-wave) ----
    float bm[8];
    #pragma unroll
    for (int v = 0; v < 8; ++v) {
      float t = s[0][v];
      #pragma unroll
      for (int ct = 1; ct < 8; ++ct) t = fmaxf(t, s[ct][v]);
      t = fmaxf(t, __shfl_xor(t, 1, 32));
      t = fmaxf(t, __shfl_xor(t, 2, 32));
      t = fmaxf(t, __shfl_xor(t, 4, 32));
      t = fmaxf(t, __shfl_xor(t, 8, 32));
      bm[v] = t;
    }
    float corr[8];
    #pragma unroll
    for (int v = 0; v < 8; ++v) {
      const float mn = fmaxf(mrun[v], bm[v]);
      corr[v] = __expf(mrun[v] - mn);
      mrun[v] = mn;
      lrun[v] *= corr[v];
    }
    #pragma unroll
    for (int i = 0; i < 4; ++i)
      #pragma unroll
      for (int v = 0; v < 8; ++v) oacc[i][v] *= corr[v];

    // ---- P = exp(S - m), row sums, stash P tile (own wave region) ----
    float rs[8];
    #pragma unroll
    for (int v = 0; v < 8; ++v) rs[v] = 0.0f;
    #pragma unroll
    for (int ct = 0; ct < 8; ++ct) {
      #pragma unroll
      for (int v = 0; v < 8; ++v) {
        const float p = __expf(s[ct][v] - mrun[v]);
        rs[v] += p;
        Pl[wave][v + 8 * hg][ct * 16 + m] = f2bf(p);
      }
    }
    #pragma unroll
    for (int v = 0; v < 8; ++v) {
      float t = rs[v];
      t += __shfl_xor(t, 1, 32);
      t += __shfl_xor(t, 2, 32);
      t += __shfl_xor(t, 4, 32);
      t += __shfl_xor(t, 8, 32);
      lrun[v] += t;
    }

    __syncthreads();   // P stores complete before A-layout reads

    // ---- O += P @ V^T  (4 c-out tiles, K=128 in 4 chained WMMAs) ----
    const uint32_t* pb = (const uint32_t*)&Pl[wave][m][0];   // row stride 66 dwords
    #pragma unroll
    for (int kt = 0; kt < 4; ++kt) {
      BF16x16 pa;
      #pragma unroll
      for (int j = 0; j < 8; ++j)
        pa.u[j] = pb[kt * 16 + hg * 8 + j];
      #pragma unroll
      for (int i = 0; i < 4; ++i) {
        const uint32_t* vp = Vs32 + (size_t)(i * 16 + m) * 68 + kt * 16;
        BF16x16 vb;
        #pragma unroll
        for (int j = 0; j < 8; ++j) vb.u[j] = vp[hg * 8 + j];
        oacc[i] = WMMA_BF16(pa.v, vb.v, oacc[i]);
      }
    }
  }

  // ---- normalize by row sum and store At[b][t][c] ----
  const int b = bh >> 3, head = bh & 7;
  #pragma unroll
  for (int v = 0; v < 8; ++v) {
    const float inv = 1.0f / lrun[v];
    const int tq = qrow0 + v + 8 * hg;
    #pragma unroll
    for (int i = 0; i < 4; ++i) {
      const int c = head * 64 + i * 16 + m;
      At[((size_t)b * 1024 + tq) * 512 + c] = f2bf(oacc[i][v] * inv);
    }
  }
}

// ---------------------------------------------------------------------------
// Kernel 5: proj GEMM + bias + gate*out + residual (fp32 output)
// ---------------------------------------------------------------------------
__global__ void __launch_bounds__(256)
k_proj(const uint16_t* __restrict__ Wp, const uint16_t* __restrict__ At,
       const float* __restrict__ proj_b, const float* __restrict__ mod,
       const float* __restrict__ x, float* __restrict__ out) {
  __shared__ uint16_t Xs[128][40];
  uint32_t* Xs32 = (uint32_t*)&Xs[0][0];
  const int b = blockIdx.z;
  const int wave = threadIdx.x >> 5, lane = threadIdx.x & 31;
  const int m = lane & 15, hg = lane >> 4;
  const int obase = blockIdx.y * 128 + wave * 16;
  const int tbase = blockIdx.x * 128;
  const uint32_t* wrow = (const uint32_t*)(Wp + (size_t)(obase + m) * 512);
  const uint4* AG4 = (const uint4*)(At + ((size_t)b * 1024 + tbase) * 512);

  v8f acc[8];
  #pragma unroll
  for (int ct = 0; ct < 8; ++ct) acc[ct] = v8zero();

  const int row0 = threadIdx.x >> 2, q0 = threadIdx.x & 3;

  for (int kk = 0; kk < 512; kk += 32) {
    __syncthreads();
    #pragma unroll
    for (int it = 0; it < 2; ++it) {
      const int row = row0 + it * 64, q = q0;
      async_g2l_b128(&Xs32[row * 20 + q * 4], &AG4[row * 64 + (kk >> 3) + q]);
    }
    const int kpf = (kk + 32 < 512) ? (kk + 32) : kk;
    __builtin_prefetch((const void*)&AG4[row0 * 64 + (kpf >> 3) + q0], 0, 1);
    wait_async0();
    __syncthreads();

    BF16x16 a;
    #pragma unroll
    for (int j = 0; j < 8; ++j)
      a.u[j] = wrow[(kk >> 1) + hg * 8 + j];
    #pragma unroll
    for (int ct = 0; ct < 8; ++ct) {
      BF16x16 bb;
      #pragma unroll
      for (int j = 0; j < 8; ++j)
        bb.u[j] = Xs32[(ct * 16 + m) * 20 + hg * 8 + j];
      acc[ct] = WMMA_BF16(a.v, bb.v, acc[ct]);
    }
  }

  #pragma unroll
  for (int ct = 0; ct < 8; ++ct) {
    const int t = tbase + ct * 16 + m;
    #pragma unroll
    for (int v = 0; v < 8; ++v) {
      const int o = obase + v + 8 * hg;
      const float r = acc[ct][v] + proj_b[o];
      const float g = mod[b * 1536 + 1024 + o];
      const size_t xi = ((size_t)b * 512 + o) * 1024 + t;
      out[xi] = x[xi] + g * r;
    }
  }
}

// ---------------------------------------------------------------------------
// Host launcher
// ---------------------------------------------------------------------------
extern "C" void kernel_launch(void* const* d_in, const int* in_sizes, int n_in,
                              void* d_out, int out_size, void* d_ws, size_t ws_size,
                              hipStream_t stream) {
  (void)in_sizes; (void)n_in; (void)out_size; (void)ws_size;
  const float* x      = (const float*)d_in[0];
  const float* emb    = (const float*)d_in[1];
  const float* qkv_w  = (const float*)d_in[2];
  const float* qkv_b  = (const float*)d_in[3];
  const float* ada_w  = (const float*)d_in[4];
  const float* ada_b  = (const float*)d_in[5];
  const float* proj_w = (const float*)d_in[6];
  const float* proj_b = (const float*)d_in[7];
  float* out = (float*)d_out;
  char*  ws  = (char*)d_ws;

  float*    mod = (float*)(ws + OFF_MOD);
  uint16_t* Wq  = (uint16_t*)(ws + OFF_WQ);
  uint16_t* Wp  = (uint16_t*)(ws + OFF_WP);
  uint16_t* XMt = (uint16_t*)(ws + OFF_XMT);
  uint16_t* Qt  = (uint16_t*)(ws + OFF_QT);
  uint16_t* Kt  = (uint16_t*)(ws + OFF_KT);
  uint16_t* Vm  = (uint16_t*)(ws + OFF_V);
  uint16_t* At  = (uint16_t*)(ws + OFF_AT);

  // 0) weights -> bf16
  k_convert<<<4096, 256, 0, stream>>>(qkv_w, proj_w, Wq, Wp);
  // 1) adaLN modulation vector
  k_mod<<<NB, 256, 0, stream>>>(emb, ada_w, ada_b, mod);
  // 2) GroupNorm + modulate -> XMt (bf16, t-major)
  k_gn<<<dim3(32, NB), 256, 0, stream>>>(x, mod, XMt);
  // 3) QKV GEMM -> Qt/Kt/V
  k_qkv<<<dim3(8, 12, NB), 256, 0, stream>>>(Wq, XMt, qkv_b, Qt, Kt, Vm);
  // 4) flash attention per head -> At  (64 q-rows per WG, 4 waves)
  k_attn<<<dim3(16, NB * NH), 128, 0, stream>>>(Qt, Kt, Vm, At);
  // 5) proj + gate + residual -> out (fp32)
  k_proj<<<dim3(8, 4, NB), 256, 0, stream>>>(Wp, At, proj_b, mod, x, out);
}
